// SegmentationAttentionSeparateModule_65223373357227
// MI455X (gfx1250) — matrix operand order repaired and checked by reading, not compiled, not run on hardware
//
#include <hip/hip_runtime.h>

// Problem constants (from reference): B=8, Dk=128, Dv=512, T=2, H=W=56
#define B_   8
#define DK   128
#define DV   512
#define M_   3136            // H*W query positions
#define N_   6272            // T*H*W memory positions
#define NC   64              // memory chunk per iteration
#define MT   16              // query tile per workgroup
#define P_SCALAR 40.0f
#define NEG_INF  -1e30f

// LDS strides (in 16-bit elements), padded; all WMMA fragment loads stay 16B aligned
#define QS_STRIDE 136        // [m][d]  16 x 128
#define KS_STRIDE 136        // [n][d]  64 x 128
#define PS_STRIDE 72         // [m][n]  16 x 64

typedef __attribute__((ext_vector_type(16))) __bf16       v16bf;
typedef __attribute__((ext_vector_type(8)))  float        v8f;
typedef __attribute__((ext_vector_type(4)))  unsigned int u32x4;
typedef __attribute__((ext_vector_type(4)))  float        f32x4;

union ABFrag { u32x4 q[2]; v16bf v; };

// Packed f32x2 -> bf16x2, round-to-nearest (+0x8000 then take high halves).
// One v_add per value + one v_perm_b32 to assemble the pair.
__device__ __forceinline__ unsigned int f2bf_pk(float x, float y) {
  unsigned int ux = __float_as_uint(x) + 0x8000u;
  unsigned int uy = __float_as_uint(y) + 0x8000u;
#if __has_builtin(__builtin_amdgcn_perm)
  // result bytes: {uy[3],uy[2],ux[3],ux[2]} -> low16 = x_hi, high16 = y_hi
  return __builtin_amdgcn_perm(uy, ux, 0x07060302u);
#else
  return (ux >> 16) | (uy & 0xffff0000u);
#endif
}

__global__ __launch_bounds__(128)
void SegmentationAttentionSeparateModule_65223373357227_kernel(
    const float* __restrict__ qkey, const unsigned char* __restrict__ qmask,
    const float* __restrict__ mkey, const float* __restrict__ mval,
    const unsigned char* __restrict__ mmask, float* __restrict__ out)
{
  __shared__ __align__(16) unsigned short Qs[MT * QS_STRIDE];
  __shared__ __align__(16) unsigned short Ks[NC * KS_STRIDE];
  __shared__ __align__(16) unsigned short Ps[MT * PS_STRIDE];
  __shared__ float smm[NC];
  __shared__ float redMax[4][16];
  __shared__ float redSum[4][16];

  const int t     = threadIdx.x;
  const int lane  = t & 31;
  const int w     = t >> 5;                 // wave id 0..3
  const int mcol  = lane & 15;              // C/D layout: N (= query col) is lane%16
  const int hi    = (lane >= 16) ? 8 : 0;   // A-fragment K interleave per ISA 7.12.2
  const int sel16 = (lane >= 16) ? 16 : 0;  // unused; B-fragments use the same hi interleave as A

  const int b  = blockIdx.x / (M_ / MT);
  const int mt = blockIdx.x % (M_ / MT);
  const int m0 = mt * MT;

  const float* qk_b = qkey + (size_t)b * DK * M_;
  const float* mk_b = mkey + (size_t)b * DK * N_;
  const float* mv_b = mval + (size_t)b * DV * N_;
  const unsigned char* mm_b = mmask + (size_t)b * N_;
  const float qmf = qmask[(size_t)b * M_ + m0 + mcol] ? 1.0f : 0.0f;

  // Stage Q tile into LDS as bf16 [m][d], folding the 40x score scale into Q.
  // Packed over d pairs: 1024 pairs = 64 dp x 16 m.
  for (int i = 0; i < 8; ++i) {
    int idx = i * 128 + t;
    int dp  = idx >> 4;
    int m   = idx & 15;
    float x = qk_b[(size_t)(2 * dp)     * M_ + m0 + m] * P_SCALAR;
    float y = qk_b[(size_t)(2 * dp + 1) * M_ + m0 + m] * P_SCALAR;
    *(unsigned int*)&Qs[m * QS_STRIDE + 2 * dp] = f2bf_pk(x, y);
  }
  __syncthreads();

  v8f acc[8];                               // 128 Dv rows per wave x 16 m, f32
  #pragma unroll
  for (int i = 0; i < 8; ++i)
    #pragma unroll
    for (int r = 0; r < 8; ++r) acc[i][r] = 0.0f;
  float rmax = NEG_INF;                     // running max for m = mcol
  float rsum = 0.0f;                        // running denom for m = mcol

  for (int n0 = 0; n0 < N_; n0 += NC) {
    __syncthreads();                        // previous-iteration readers done

    if (n0 + NC < N_) {                     // CDNA5 global_prefetch_b8
      __builtin_prefetch(&mk_b[(size_t)t * N_ + n0 + NC], 0, 1);
      __builtin_prefetch(&mv_b[(size_t)(t * 4) * N_ + n0 + NC], 0, 1);
    }

    // Stage K chunk as bf16 [n][d] (transpose of global [d][n]); coalesced reads,
    // packed over d pairs: 4096 pairs = 64 dp x 64 n.
    for (int i = 0; i < 32; ++i) {
      int idx = i * 128 + t;
      int dp  = idx >> 6;
      int n   = idx & 63;
      float x = mk_b[(size_t)(2 * dp)     * N_ + n0 + n];
      float y = mk_b[(size_t)(2 * dp + 1) * N_ + n0 + n];
      *(unsigned int*)&Ks[n * KS_STRIDE + 2 * dp] = f2bf_pk(x, y);
    }
    if (t < NC) smm[t] = mm_b[n0 + t] ? 1.0f : 0.0f;
    __syncthreads();

    // ---- GEMM1: S(16n x 16m) = Kchunk_tile * Qtile, per wave ----
    v8f c;
    #pragma unroll
    for (int r = 0; r < 8; ++r) c[r] = 0.0f;
    const int nrow = w * 16 + mcol;         // A-fragment M row = lane%16
    #pragma unroll
    for (int kk = 0; kk < DK; kk += 32) {
      ABFrag a, bf;
      a.q[0]  = *(const u32x4*)&Ks[nrow * KS_STRIDE + kk + hi];
      a.q[1]  = *(const u32x4*)&Ks[nrow * KS_STRIDE + kk + 16 + hi];
      bf.q[0] = *(const u32x4*)&Qs[mcol * QS_STRIDE + kk + hi];
      bf.q[1] = *(const u32x4*)&Qs[mcol * QS_STRIDE + kk + 16 + hi];
      c = __builtin_amdgcn_wmma_f32_16x16x32_bf16(false, a.v, false, bf.v,
                                                  (short)0, c, false, false);
    }

    // ---- Online softmax over this chunk (per query column m = mcol) ----
    float pr[8], vm[8];
    float cmax = NEG_INF;
    #pragma unroll
    for (int r = 0; r < 8; ++r) {
      int nl = w * 16 + r + hi;             // D layout: M row = r (+8 for hi half)
      vm[r] = smm[nl];
      float s = (vm[r] != 0.0f) ? c[r] : NEG_INF;
      pr[r] = s;
      cmax = fmaxf(cmax, s);
    }
    cmax = fmaxf(cmax, __shfl_xor(cmax, 16, 32));   // combine lane halves (same m)
    if (lane < 16) redMax[w][lane] = cmax;
    __syncthreads();
    float nmax = rmax;
    for (int ww = 0; ww < 4; ++ww) nmax = fmaxf(nmax, redMax[ww][mcol]);
    float alpha = __expf(rmax - nmax);
    rmax = nmax;

    float csum = 0.0f;
    #pragma unroll
    for (int r = 0; r < 8; ++r) {
      float p = (vm[r] != 0.0f) ? __expf(pr[r] - nmax) : 0.0f;  // masked rows -> 0
      pr[r] = p;
      csum += p;
    }
    csum += __shfl_xor(csum, 16, 32);
    if (lane < 16) redSum[w][lane] = csum;

    // Write P tile to LDS as bf16 [m][n]; lane's 8 values are consecutive n, so
    // pack pairs (aligned, since w*16+hi is a multiple of 8).
    #pragma unroll
    for (int r = 0; r < 8; r += 2) {
      *(unsigned int*)&Ps[mcol * PS_STRIDE + w * 16 + hi + r] = f2bf_pk(pr[r], pr[r + 1]);
    }
    __syncthreads();
    float chunkSum = 0.0f;
    for (int ww = 0; ww < 4; ++ww) chunkSum += redSum[ww][mcol];
    rsum = rsum * alpha + chunkSum;

    // Rescale persistent O accumulators (alpha is per-lane since m = lane%16).
    #pragma unroll
    for (int i = 0; i < 8; ++i)
      #pragma unroll
      for (int r = 0; r < 8; ++r) acc[i][r] *= alpha;

    // ---- GEMM2: O += V * P. V has no intra-workgroup reuse, so build the
    // A-fragment directly from global memory (n is contiguous there): two 16B
    // f32 loads per 8-wide K run, packed-convert to bf16 in registers. ----
    #pragma unroll
    for (int ai = 0; ai < 8; ++ai) {
      int dvc = ai >> 1, j = ai & 1;
      int dvrow = dvc * 128 + w * 32 + j * 16 + mcol;       // A-fragment M row
      const float* vrow = mv_b + (size_t)dvrow * N_ + n0;
      #pragma unroll
      for (int kk2 = 0; kk2 < NC; kk2 += 32) {
        f32x4 a0 = *(const f32x4*)(vrow + kk2 + hi);
        f32x4 a1 = *(const f32x4*)(vrow + kk2 + hi + 4);
        f32x4 a2 = *(const f32x4*)(vrow + kk2 + 16 + hi);
        f32x4 a3 = *(const f32x4*)(vrow + kk2 + 16 + hi + 4);
        ABFrag a, bf;
        a.q[0][0] = f2bf_pk(a0.x, a0.y);
        a.q[0][1] = f2bf_pk(a0.z, a0.w);
        a.q[0][2] = f2bf_pk(a1.x, a1.y);
        a.q[0][3] = f2bf_pk(a1.z, a1.w);
        a.q[1][0] = f2bf_pk(a2.x, a2.y);
        a.q[1][1] = f2bf_pk(a2.z, a2.w);
        a.q[1][2] = f2bf_pk(a3.x, a3.y);
        a.q[1][3] = f2bf_pk(a3.z, a3.w);
        bf.q[0] = *(const u32x4*)&Ps[mcol * PS_STRIDE + kk2 + hi];
        bf.q[1] = *(const u32x4*)&Ps[mcol * PS_STRIDE + kk2 + 16 + hi];
        acc[ai] = __builtin_amdgcn_wmma_f32_16x16x32_bf16(false, a.v, false, bf.v,
                                                          (short)0, acc[ai], false, false);
      }
    }
  }

  // Finalize: divide by softmax denom, apply query mask (and all-masked guard).
  const float inv = (qmf != 0.0f && rsum > 0.0f) ? (1.0f / rsum) : 0.0f;
  #pragma unroll
  for (int ai = 0; ai < 8; ++ai) {
    int dvc = ai >> 1, j = ai & 1;
    int dvbase = dvc * 128 + w * 32 + j * 16 + hi;
    #pragma unroll
    for (int r = 0; r < 8; ++r) {
      int dv = dvbase + r;
      out[((size_t)b * DV + dv) * M_ + m0 + mcol] = acc[ai][r] * inv;
    }
  }
}

extern "C" void kernel_launch(void* const* d_in, const int* in_sizes, int n_in,
                              void* d_out, int out_size, void* d_ws, size_t ws_size,
                              hipStream_t stream) {
  const float*         qkey  = (const float*)d_in[0];
  const unsigned char* qmask = (const unsigned char*)d_in[1];
  const float*         mkey  = (const float*)d_in[2];
  const float*         mval  = (const float*)d_in[3];
  const unsigned char* mmask = (const unsigned char*)d_in[4];
  float* out = (float*)d_out;

  dim3 grid(B_ * (M_ / MT));   // 8 * 196 = 1568 workgroups
  dim3 block(128);             // 4 waves (wave32)
  SegmentationAttentionSeparateModule_65223373357227_kernel<<<grid, block, 0, stream>>>(
      qkey, qmask, mkey, mval, mmask, out);
}
